// SCDMNSlicedReg_23270132810072
// MI455X (gfx1250) — matrix operand order repaired and checked by reading, not compiled, not run on hardware
//
#include <hip/hip_runtime.h>
#include <stdint.h>

// ---------------------------------------------------------------------------
// Sliced-ResNet forward for MI455X (gfx1250), bf16 WMMA implicit-GEMM convs.
//
// Roofline: ~590 GFLOP vs ~2-3 GB HBM traffic at 23.3 TB/s -> must run convs
// on the matrix pipes. v_wmma_f32_16x16x32_bf16 (f32 accumulate).
// A (weight) tiles: TDM tensor_load_to_lds (3x3 convs, TENSORcnt) or
// global_load_async_to_lds_b128 (1x1 convs, ASYNCcnt). B (im2col, masked)
// staged via registers. Double-buffered LDS, software-pipelined; fused
// BN-fold / residual / ReLU / per-sample-mask epilogue.
// All pixel decodes use power-of-two shifts (H,W in {32,16,8,4}).
// ---------------------------------------------------------------------------

typedef __bf16 bf16_t;
typedef bf16_t bf16x16 __attribute__((ext_vector_type(16)));
typedef float  f32x8   __attribute__((ext_vector_type(8)));
typedef uint32_t u32x4 __attribute__((ext_vector_type(4)));
typedef uint32_t u32x8 __attribute__((ext_vector_type(8)));

union ABFrag { bf16x16 v; uint32_t u[8]; };

__device__ __forceinline__ uint16_t f2bf(float f) {
  union { float f; uint32_t u; } x; x.f = f;
  uint32_t r = x.u + 0x7FFFu + ((x.u >> 16) & 1u);  // round-to-nearest-even
  return (uint16_t)(r >> 16);
}

// ---- BN fold: scale = g/sqrt(v+eps), bias = b - m*scale --------------------
__global__ void bn_fold_kernel(const float* __restrict__ g,
                               const float* __restrict__ b,
                               const float* __restrict__ m,
                               const float* __restrict__ v,
                               float* __restrict__ scale,
                               float* __restrict__ bias, int C) {
  int c = blockIdx.x * blockDim.x + threadIdx.x;
  if (c < C) {
    float s = g[c] * rsqrtf(v[c] + 1e-5f);
    scale[c] = s;
    bias[c]  = b[c] - m[c] * s;
  }
}

// ---- per-sample soft mask: out[b,c] = sigmoid(scores[ctx[b], c]) -----------
__global__ void mask_kernel(const float* __restrict__ scores,
                            const int* __restrict__ ctx,
                            float* __restrict__ out, int C, int n) {
  int i = blockIdx.x * blockDim.x + threadIdx.x;
  if (i < n) {
    int b = i / C, c = i - b * C;
    float s = scores[ctx[b] * C + c];
    out[i] = 1.0f / (1.0f + expf(-s));
  }
}

// ---- weight f32 -> bf16 bits ----------------------------------------------
__global__ void w_to_bf16_kernel(const float* __restrict__ w,
                                 uint16_t* __restrict__ wb, int n) {
  int i = blockIdx.x * blockDim.x + threadIdx.x;
  if (i < n) wb[i] = f2bf(w[i]);
}

// ---- implicit-GEMM conv + fused BN / residual / relu / mask ----------------
// GEMM: M=Cout (grid.y*64), N=B*Ho*Wo (grid.x*128), K=Cin*KH*KW.
// 8 wave32 / WG; wave (mi,ni) owns 16 M-rows x 64 N-cols (4 wmma tiles).
template <int KH, int KW>
__global__ __launch_bounds__(256) void conv_wmma_kernel(
    const uint16_t* __restrict__ Wb,    // bf16 weights [Cout, Cin*KH*KW]
    const float* __restrict__ X,        // [B, Cin, Hin, Win]
    float* __restrict__ Y,              // [B, Cout, Ho, Wo]
    const float* __restrict__ scale,    // [Cout]
    const float* __restrict__ bias,     // [Cout]
    const float* __restrict__ m_in,     // [B, Cin] or null
    const float* __restrict__ m_out,    // [B, Cout] or null
    const float* __restrict__ res,      // [B, Cout, Ho, Wo] or null
    const float* __restrict__ res_mask, // [B, Cout] or null
    int Cin, int Hin, int Win, int Cout,
    int woSh, int hwSh, int stride, int pad, int relu, int alignedA) {
  constexpr int KHW = KH * KW;
  __shared__ alignas(16) uint16_t Ash[2][64 * 32];   // [buf][row(co)*32 + k]
  __shared__ alignas(16) uint16_t Bsh[2][128 * 32];  // [buf][col(pix)*32 + k]

  const int K    = Cin * KHW;
  const int tid  = threadIdx.x;
  const int lane = tid & 31;
  const int wave = tid >> 5;
  const int mi   = wave & 3;      // M subtile (16 rows)
  const int ni   = wave >> 2;     // N half (64 cols)
  const int rowL = lane & 15;
  const int g    = lane >> 4;
  const int coBase = blockIdx.y * 64;
  const int nBase  = blockIdx.x * 128;
  const int hwMask = (1 << hwSh) - 1;
  const int woMask = (1 << woSh) - 1;

  // --- per-thread loop-invariant staging coordinates ---
  // B stage: e = tid + 256*i  =>  col = tid&127 (const), kk = (tid>>7) + 2*i
  const int colT = tid & 127;
  const int kkB0 = tid >> 7;
  const int nT   = nBase + colT;
  const int bT   = nT >> hwSh;
  const int rT   = nT & hwMask;
  const int iy0  = (rT >> woSh) * stride - pad;
  const int ix0  = (rT & woMask) * stride - pad;
  const long xRow = (long)bT * Cin;
  // A stage (scalar fallback): kk = tid&31 (const), row = (tid>>5) + 8*i
  const int kkA   = tid & 31;
  const int rowA0 = tid >> 5;

  auto loadB = [&](int k0, uint16_t* bReg) {
#pragma unroll
    for (int i = 0; i < 16; ++i) {
      int k = k0 + kkB0 + 2 * i;
      uint16_t v = 0;
      if (k < K) {
        int ci = k / KHW;            // constant divisor (9 or 1)
        int tap = k - ci * KHW;
        int ky = (KH == 1) ? 0 : (tap / KW);
        int kx = (KW == 1) ? 0 : (tap - ky * KW);
        int iy = iy0 + ky, ix = ix0 + kx;
        if ((unsigned)iy < (unsigned)Hin && (unsigned)ix < (unsigned)Win) {
          float xv = X[((xRow + ci) * Hin + iy) * Win + ix];
          if (m_in) xv *= m_in[bT * Cin + ci];
          v = f2bf(xv);
        }
      }
      bReg[i] = v;
    }
  };
  auto storeB = [&](int buf, const uint16_t* bReg) {
#pragma unroll
    for (int i = 0; i < 16; ++i)
      Bsh[buf][colT * 32 + kkB0 + 2 * i] = bReg[i];
  };
  auto issueA = [&](int k0, int buf) {
    if (alignedA) {
      if (KH == 3) {
        // TDM: one DMA descriptor per WG moves the 64x32 bf16 A tile.
        // D# group0/group1 bit layout per CDNA5 ISA.
        if (wave == 0) {
          uint64_t ga =
              (uint64_t)(uintptr_t)(Wb + (size_t)coBase * K + k0);
          uint32_t lds = (uint32_t)(uintptr_t)&Ash[buf][0];
          u32x4 g0;
          g0[0] = 1u;                                   // count=1, load D#
          g0[1] = lds;                                  // lds_addr
          g0[2] = (uint32_t)ga;                         // global_addr lo
          g0[3] = (uint32_t)(ga >> 32) | (2u << 30);    // addr hi | type=2
          u32x8 g1;
          g1[0] = 1u << 16;                             // data_size=2B
          g1[1] = ((uint32_t)K & 0xFFFFu) << 16;        // tensor_dim0 lo16
          g1[2] = (((uint32_t)K >> 16) & 0xFFFFu) |
                  (((uint32_t)Cout & 0xFFFFu) << 16);   // dim0 hi | dim1 lo
          g1[3] = (((uint32_t)Cout >> 16) & 0xFFFFu) |
                  (32u << 16);                          // dim1 hi | tile_dim0
          g1[4] = 64u;                                  // tile_dim1 (dim2=0)
          g1[5] = (uint32_t)K;                          // dim0_stride lo32
          g1[6] = 0u;                                   // stride hi | d1s lo
          g1[7] = 0u;
          asm volatile("tensor_load_to_lds %0, %1"
                       :: "s"(g0), "s"(g1) : "memory");
        }
      } else {
        // 1x1 convs: per-thread 16B async copy (ASYNCcnt tracked)
        int e16 = tid * 8;
        int row = e16 >> 5, kk = e16 & 31;
        uint32_t lds = (uint32_t)(uintptr_t)&Ash[buf][e16];
        uint64_t ga =
            (uint64_t)(uintptr_t)(Wb + (long)(coBase + row) * K + k0 + kk);
        asm volatile("global_load_async_to_lds_b128 %0, %1, off"
                     :: "v"(lds), "v"(ga) : "memory");
      }
    } else {  // stem conv (K=27): scalar zero-padded path
#pragma unroll
      for (int i = 0; i < 8; ++i) {
        int row = rowA0 + 8 * i;
        int k = k0 + kkA;
        uint16_t v = (k < K) ? Wb[(long)(coBase + row) * K + k] : (uint16_t)0;
        Ash[buf][row * 32 + kkA] = v;
      }
    }
  };
  auto waitA = [&]() {
    if (alignedA) {
      if (KH == 3) __builtin_amdgcn_s_wait_tensorcnt(0);
      else asm volatile("s_wait_asynccnt 0" ::: "memory");
    }
  };

  const f32x8 vz = {0.f, 0.f, 0.f, 0.f, 0.f, 0.f, 0.f, 0.f};
  f32x8 acc[4];
  acc[0] = vz; acc[1] = vz; acc[2] = vz; acc[3] = vz;

  const int kChunks = (K + 31) >> 5;
  uint16_t bReg[16];
  issueA(0, 0);
  loadB(0, bReg);
  for (int kc = 0; kc < kChunks; ++kc) {
    const int buf = kc & 1;
    storeB(buf, bReg);
    waitA();                         // A tile landed in LDS (this wave's cnt)
    __syncthreads();
    if (kc + 1 < kChunks) {          // software pipeline: next chunk in flight
      issueA((kc + 1) << 5, buf ^ 1);
      loadB((kc + 1) << 5, bReg);
    }

    // A fragment: CDNA5 16-bit A 16x32 layout (K pairs across 8 VGPRs)
    ABFrag a;
    const uint32_t* Ap =
        (const uint32_t*)(&Ash[buf][(mi * 16 + rowL) * 32]);
#pragma unroll
    for (int v = 0; v < 8; ++v)
      a.u[v] = Ap[((v < 4) ? 0 : 8) + g * 4 + (v & 3)];

#pragma unroll
    for (int t = 0; t < 4; ++t) {
      ABFrag bf;  // B fragment: lane = N col, K halves by lane group
      const uint32_t* Bp =
          (const uint32_t*)(&Bsh[buf][(ni * 64 + t * 16 + rowL) * 32]);
#pragma unroll
      for (int v = 0; v < 8; ++v) bf.u[v] = Bp[((v < 4) ? 0 : 8) + g * 4 + (v & 3)];
      acc[t] = __builtin_amdgcn_wmma_f32_16x16x32_bf16(
          false, a.v, false, bf.v, (short)0, acc[t], false, false);
    }
    __syncthreads();
  }

  // Epilogue: D layout VGPR v -> M = v + 8*(lane/16), N = lane%16.
  // Output index is linear in r = n & (HoWo-1); no oy/ox needed.
  const int co0 = coBase + mi * 16;
#pragma unroll
  for (int t = 0; t < 4; ++t) {
    int n = nBase + ni * 64 + t * 16 + rowL;
    int b = n >> hwSh;
    int r = n & hwMask;
#pragma unroll
    for (int v = 0; v < 8; ++v) {
      int co = co0 + v + g * 8;
      float y = acc[t][v] * scale[co] + bias[co];
      long idx = (((long)b * Cout + co) << hwSh) + r;
      if (res) {
        float rv = res[idx];
        if (res_mask) rv *= res_mask[b * Cout + co];
        y += rv;
      }
      if (relu) y = fmaxf(y, 0.f);
      if (m_out) y *= m_out[b * Cout + co];
      Y[idx] = y;
    }
  }
}

// ---- fused avgpool(4x4) + masked FC + tanh, one block per sample -----------
__global__ __launch_bounds__(256) void pool_fc_kernel(
    const float* __restrict__ X,   // [B, C, HW]
    const float* __restrict__ m,   // [B, C]
    const float* __restrict__ fcw, // [C]
    const float* __restrict__ fcb, // [1]
    float* __restrict__ out, int C, int HW) {
  __shared__ float red[256];
  int b = blockIdx.x, tid = threadIdx.x;
  float s = 0.f;
  for (int c = tid; c < C; c += 256) {
    const float* p = X + ((long)b * C + c) * HW;
    float a = 0.f;
    for (int i = 0; i < HW; ++i) a += p[i];
    s += (a / (float)HW) * m[b * C + c] * fcw[c];
  }
  red[tid] = s;
  __syncthreads();
  for (int off = 128; off > 0; off >>= 1) {
    if (tid < off) red[tid] += red[tid + off];
    __syncthreads();
  }
  if (tid == 0) out[b] = tanhf(red[0] + fcb[0]);
}

// ---------------------------------------------------------------------------
extern "C" void kernel_launch(void* const* d_in, const int* in_sizes, int n_in,
                              void* d_out, int out_size, void* d_ws,
                              size_t ws_size, hipStream_t stream) {
  (void)in_sizes; (void)n_in; (void)out_size; (void)ws_size;
  const int B = 256;
  static const int SCH[4] = {64, 128, 256, 512};
  static const int SBL[4] = {3, 4, 6, 3};

  // --- parse inputs in setup_inputs() insertion order ---
  int ip = 0;
  auto nf = [&]() -> const float* { return (const float*)d_in[ip++]; };
  const float* x = nf();
  const int* ctx = (const int*)d_in[ip++];
  const float* conv1_w = nf();
  const float* bn1p[4];
  for (int i = 0; i < 4; ++i) bn1p[i] = nf();

  struct Blk {
    const float* w1; const float* b1[4];
    const float* w2; const float* b2[4];
    const float* wd; const float* bd[4];
  };
  Blk blks[16];
  int nb = 0, in_p = 64;
  for (int si = 0; si < 4; ++si) {
    int planes = SCH[si];
    int stride0 = (si == 0) ? 1 : 2;
    for (int bj = 0; bj < SBL[si]; ++bj) {
      int s = (bj == 0) ? stride0 : 1;
      Blk& bk = blks[nb++];
      bk.w1 = nf(); for (int i = 0; i < 4; ++i) bk.b1[i] = nf();
      bk.w2 = nf(); for (int i = 0; i < 4; ++i) bk.b2[i] = nf();
      if (s != 1 || in_p != planes) {
        bk.wd = nf(); for (int i = 0; i < 4; ++i) bk.bd[i] = nf();
      } else {
        bk.wd = nullptr;
        for (int i = 0; i < 4; ++i) bk.bd[i] = nullptr;
      }
      in_p = planes;
    }
  }
  const float* fc_w = nf();
  const float* fc_b = nf();
  const float* scores[4];
  for (int i = 0; i < 4; ++i) scores[i] = nf();

  // --- workspace carve ---
  uintptr_t p = (uintptr_t)d_ws;
  auto carve = [&](size_t bytes) -> void* {
    void* r = (void*)p;
    p += (bytes + 255) & ~(size_t)255;
    return r;
  };
  float* act[4];
  for (int i = 0; i < 4; ++i)
    act[i] = (float*)carve((size_t)B * 64 * 32 * 32 * sizeof(float));
  uint16_t* wbuf = (uint16_t*)carve((size_t)512 * 512 * 9 * sizeof(uint16_t));
  float* scbuf = (float*)carve(512 * sizeof(float));
  float* bibuf = (float*)carve(512 * sizeof(float));
  float* masks[4];
  for (int i = 0; i < 4; ++i)
    masks[i] = (float*)carve((size_t)B * SCH[i] * sizeof(float));

  // --- per-context soft channel masks ---
  for (int si = 0; si < 4; ++si) {
    int n = B * SCH[si];
    mask_kernel<<<(n + 255) / 256, 256, 0, stream>>>(scores[si], ctx,
                                                     masks[si], SCH[si], n);
  }

  auto run_conv = [&](const float* w, const float* const* bn, const float* Xp,
                      float* Yp, const float* min_, const float* mout_,
                      const float* resp, const float* resm, int Cin, int Hin,
                      int Cout, int Ho, int KH, int stride, int relu) {
    int K = Cin * KH * KH, nW = Cout * K;
    bn_fold_kernel<<<(Cout + 255) / 256, 256, 0, stream>>>(
        bn[0], bn[1], bn[2], bn[3], scbuf, bibuf, Cout);
    w_to_bf16_kernel<<<(nW + 255) / 256, 256, 0, stream>>>(w, wbuf, nW);
    int woSh = __builtin_ctz((unsigned)Ho);  // Wo == Ho, power of two
    int hwSh = 2 * woSh;
    int alignedA = ((K & 7) == 0) ? 1 : 0;
    dim3 grid((unsigned)((B * Ho * Ho) / 128), (unsigned)(Cout / 64));
    if (KH == 3)
      conv_wmma_kernel<3, 3><<<grid, 256, 0, stream>>>(
          wbuf, Xp, Yp, scbuf, bibuf, min_, mout_, resp, resm, Cin, Hin, Hin,
          Cout, woSh, hwSh, stride, 1, relu, alignedA);
    else
      conv_wmma_kernel<1, 1><<<grid, 256, 0, stream>>>(
          wbuf, Xp, Yp, scbuf, bibuf, min_, mout_, resp, resm, Cin, Hin, Hin,
          Cout, woSh, hwSh, stride, 0, relu, alignedA);
  };

  // --- stem: conv1 + bn1 + relu ---
  run_conv(conv1_w, bn1p, x, act[0], nullptr, nullptr, nullptr, nullptr, 3, 32,
           64, 32, 3, 1, 1);

  // --- blocks ---
  int xi = 0, Hcur = 32;
  in_p = 64;
  nb = 0;
  for (int si = 0; si < 4; ++si) {
    int planes = SCH[si];
    int stride0 = (si == 0) ? 1 : 2;
    const float* m_cur = masks[si];
    for (int bj = 0; bj < SBL[si]; ++bj) {
      Blk& bk = blks[nb++];
      int s = (bj == 0) ? stride0 : 1;
      int Ho = Hcur / s;
      const float* m_in =
          (bj > 0) ? m_cur : ((si == 0) ? nullptr : masks[si - 1]);
      float* h1 = act[(xi + 1) & 3];
      float* scv = act[(xi + 2) & 3];
      float* ob  = act[(xi + 3) & 3];
      // h = relu(bn1(conv1(x * m_in)))   [m_mid folded into next conv]
      run_conv(bk.w1, bk.b1, act[xi], h1, m_in, nullptr, nullptr, nullptr,
               in_p, Hcur, planes, Ho, 3, s, 1);
      const float* resp; const float* resm;
      if (bk.wd) {  // sc = bnd(convd(x * m_in))
        run_conv(bk.wd, bk.bd, act[xi], scv, m_in, nullptr, nullptr, nullptr,
                 in_p, Hcur, planes, Ho, 1, s, 0);
        resp = scv; resm = nullptr;
      } else {      // sc = x * m_in (identity)
        resp = act[xi]; resm = m_in;
      }
      // out = relu(bn2(conv2(h * m_mid)) + sc) * m_out
      run_conv(bk.w2, bk.b2, h1, ob, m_cur, m_cur, resp, resm, planes, Ho,
               planes, Ho, 3, 1, 1);
      xi = (xi + 3) & 3;
      Hcur = Ho;
      in_p = planes;
    }
  }

  // --- avgpool + masked fc + tanh ---
  pool_fc_kernel<<<B, 256, 0, stream>>>(act[xi], masks[3], fc_w, fc_b,
                                        (float*)d_out, 512, 16);
}
